// BertEmbedding_4252017623405
// MI455X (gfx1250) — matrix operand order repaired and checked by reading, not compiled, not run on hardware
//
#include <hip/hip_runtime.h>
#include <hip/hip_bf16.h>
#include <math.h>

// ---------------- problem constants (match reference) ----------------
#define Bq    64
#define Sq    512
#define Dq    768
#define Hq    4
#define DHq   192          // D / H
#define Pq    8
#define NPERq 9            // 1 + P
#define Nq    576          // B * NPER
#define ERAW  4096
#define ESYM  (2*ERAW + Nq)   // 8768 (symmetrized + self loops)
#define MAXDEG 512
#define LDSPAD 40          // padded K-stride (elements) for LDS tiles

typedef __attribute__((ext_vector_type(16))) __bf16 v16bf;
typedef __attribute__((ext_vector_type(8)))  __bf16 v8bf;
typedef __attribute__((ext_vector_type(4)))  __bf16 v4bf;
typedef __attribute__((ext_vector_type(8)))  float  v8f;

// edge t -> (src, dst) of symmetrized edge list with self loops
__device__ __forceinline__ void edge_sd(const int* __restrict__ eidx, int t, int& s, int& d) {
    if (t < ERAW)            { s = eidx[t];               d = eidx[ERAW + t];     }
    else if (t < 2*ERAW)     { s = eidx[ERAW + (t-ERAW)]; d = eidx[t-ERAW];       }
    else                     { s = t - 2*ERAW;            d = s;                  }
}

__device__ __forceinline__ float gelu_exact(float x) {
    return 0.5f * x * (1.0f + erff(x * 0.70710678118654752440f));
}

// ---------------- 1) GNN node assembly: CLS row + span sums ----------------
// grid = N blocks, block = 192 threads, float4 per thread (768 floats/row)
__global__ void assemble_nodes_kernel(const int* __restrict__ src,
                                      const int* __restrict__ span_start,
                                      const int* __restrict__ span_end,
                                      const float* __restrict__ word_table,
                                      float* __restrict__ xbuf) {
    const int n = blockIdx.x;
    const int b = n / NPERq;
    const int j = n - b * NPERq;
    const int t = threadIdx.x;
    const float4* wt4 = (const float4*)word_table;   // row stride = 192 float4
    float ax = 0.f, ay = 0.f, az = 0.f, aw = 0.f;
    if (j == 0) {
        int tok = src[b * Sq];
        float4 v = wt4[(size_t)tok * (Dq/4) + t];
        ax = v.x; ay = v.y; az = v.z; aw = v.w;
    } else {
        const int p  = j - 1;
        const int s0 = span_start[b * Pq + p];
        const int s1 = span_end  [b * Pq + p];      // inclusive end
        for (int s = s0; s <= s1; ++s) {
            int tok = src[b * Sq + s];
            float4 v = wt4[(size_t)tok * (Dq/4) + t];
            ax += v.x; ay += v.y; az += v.z; aw += v.w;
        }
    }
    float4 o; o.x = ax; o.y = ay; o.z = az; o.w = aw;
    ((float4*)xbuf)[(size_t)n * (Dq/4) + t] = o;
}

// ---------------- 2) h = x @ W  via bf16 WMMA, f32 accumulate ----------------
// grid = (36, 6): 36 M-tiles of 16, 6 N-panels of 128. 256 threads = 8 waves,
// wave w owns columns [w*16, w*16+16). B is staged TRANSPOSED in LDS so each
// lane's 16-K fragment slice is contiguous -> 2x ds_load_b128 per fragment.
__global__ __launch_bounds__(256)
void gemm_wmma_kernel(const float* __restrict__ X,   // [N, D]
                      const float* __restrict__ W,   // [D, D]
                      float* __restrict__ Hout) {    // [N, D]
    __shared__ __bf16 Abf[16 * LDSPAD];    // A tile, row-major [m][k], padded
    __shared__ __bf16 Bt [128 * LDSPAD];   // B panel, transposed [col][k], padded

    const int t    = threadIdx.x;
    const int lane = t & 31;
    const int wid  = t >> 5;
    const int m0   = blockIdx.x * 16;
    const int n0   = blockIdx.y * 128;

    v8f c = {};

    const int mrow = lane & 15;
    const int col  = wid * 16 + (lane & 15);   // column within 128-panel
    const int hiA  = (lane >> 4) * 16;         // A K-half select (contiguous k = 16h + s, same as B)
    const int hiB  = (lane >> 4) * 16;         // B K-half select

    for (int k0 = 0; k0 < Dq; k0 += 32) {
        // stage A tile 16x32 (one float4 -> packed v4bf ds_store_b64)
        if (t < 128) {
            const int r   = t >> 3;
            const int cix = (t & 7) * 4;
            const float4 v = *(const float4*)(X + (size_t)(m0 + r) * Dq + k0 + cix);
            v4bf pk;
            pk[0] = (__bf16)v.x; pk[1] = (__bf16)v.y;
            pk[2] = (__bf16)v.z; pk[3] = (__bf16)v.w;
            *(v4bf*)(Abf + r * LDSPAD + cix) = pk;
        }
        // stage B panel 32x128, transposed into Bt[col][k]
        #pragma unroll
        for (int i = 0; i < 4; ++i) {
            const int idx4 = t + i * 256;          // 1024 float4 = 4096 floats
            const int r    = idx4 >> 5;            // k-row 0..31
            const int cix  = (idx4 & 31) * 4;      // col 0..124
            const float4 v = *(const float4*)(W + (size_t)(k0 + r) * Dq + n0 + cix);
            Bt[(cix + 0) * LDSPAD + r] = (__bf16)v.x;
            Bt[(cix + 1) * LDSPAD + r] = (__bf16)v.y;
            Bt[(cix + 2) * LDSPAD + r] = (__bf16)v.z;
            Bt[(cix + 3) * LDSPAD + r] = (__bf16)v.w;
        }
        __syncthreads();

        // fragments: contiguous 16B chunks -> ds_load_b128
        const v8bf* Ap = (const v8bf*)(Abf + mrow * LDSPAD);
        const v8bf a0 = Ap[hiA >> 3];           // K = hiA .. hiA+7
        const v8bf a1 = Ap[1 + (hiA >> 3)];     // K = hiA+8 .. hiA+15
        const v8bf* Bp = (const v8bf*)(Bt + col * LDSPAD + hiB);
        const v8bf b0 = Bp[0];                  // K = hiB .. hiB+7
        const v8bf b1 = Bp[1];                  // K = hiB+8 .. hiB+15
        const v16bf a = __builtin_shufflevector(a0, a1,
                          0,1,2,3,4,5,6,7,8,9,10,11,12,13,14,15);
        const v16bf b = __builtin_shufflevector(b0, b1,
                          0,1,2,3,4,5,6,7,8,9,10,11,12,13,14,15);

        c = __builtin_amdgcn_wmma_f32_16x16x32_bf16(false, a, false, b,
                                                    (short)0, c, false, false);
        __syncthreads();
    }

    // C/D layout: VGPR r -> M = r + 8*(lane>=16), N = lane&15
    const int mbase = m0 + (lane >> 4) * 8;
    const int ocol  = n0 + col;
    #pragma unroll
    for (int r = 0; r < 8; ++r)
        Hout[(size_t)(mbase + r) * Dq + ocol] = c[r];
}

// ---------------- 3) alpha_s / alpha_d per (node, head) ----------------
// grid = N blocks, 128 threads = 4 waves (one per head), wave32 reduction
__global__ void gat_alpha_kernel(const float* __restrict__ hmat,
                                 const float* __restrict__ a_s,
                                 const float* __restrict__ a_d,
                                 float* __restrict__ as_out,
                                 float* __restrict__ ad_out) {
    const int node = blockIdx.x;
    const int lane = threadIdx.x & 31;
    const int head = threadIdx.x >> 5;
    const float* hrow = hmat + (size_t)node * Dq + head * DHq;
    const float* vs = a_s + head * DHq;
    const float* vd = a_d + head * DHq;
    float ss = 0.f, sd = 0.f;
    for (int i = lane; i < DHq; i += 32) {
        float hv = hrow[i];
        ss = fmaf(hv, vs[i], ss);
        sd = fmaf(hv, vd[i], sd);
    }
    #pragma unroll
    for (int off = 16; off > 0; off >>= 1) {
        ss += __shfl_xor(ss, off, 32);
        sd += __shfl_xor(sd, off, 32);
    }
    if (lane == 0) {
        as_out[node * Hq + head] = ss;
        ad_out[node * Hq + head] = sd;
    }
}

// -------- 4) deterministic per-dst softmax attention + aggregate + GELU -----
// grid = N blocks (one per dst node), 256 threads. No atomics: fixed-order
// compaction + fixed reduction trees -> bitwise deterministic.
__global__ __launch_bounds__(256)
void gat_aggregate_kernel(const float* __restrict__ hmat,
                          const float* __restrict__ as,
                          const float* __restrict__ ad,
                          const int*   __restrict__ eidx,
                          const float* __restrict__ bias,
                          float* __restrict__ xout) {
    const int dst = blockIdx.x;
    const int t   = threadIdx.x;

    __shared__ int   counts[256];
    __shared__ int   offs[257];
    __shared__ int   srcs[MAXDEG];
    __shared__ float ev[Hq][MAXDEG];

    // pass 1: count matches in this thread's fixed strided subset
    int c = 0;
    for (int e = t; e < ESYM; e += 256) {
        int s, d; edge_sd(eidx, e, s, d);
        if (d == dst) c++;
    }
    counts[t] = c;
    __syncthreads();
    if (t == 0) {
        int run = 0;
        for (int i = 0; i < 256; ++i) { offs[i] = run; run += counts[i]; }
        offs[256] = run;
    }
    __syncthreads();
    int total = offs[256];
    if (total > MAXDEG) total = MAXDEG;

    // pass 2: fill neighbor list in deterministic (thread, occurrence) order
    const float ad0 = ad[dst*Hq+0], ad1 = ad[dst*Hq+1],
                ad2 = ad[dst*Hq+2], ad3 = ad[dst*Hq+3];
    int o = offs[t];
    for (int e = t; e < ESYM; e += 256) {
        int s, d; edge_sd(eidx, e, s, d);
        if (d == dst && o < MAXDEG) {
            srcs[o] = s;
            float v0 = as[s*Hq+0] + ad0; ev[0][o] = (v0 < 0.f) ? 0.2f*v0 : v0;
            float v1 = as[s*Hq+1] + ad1; ev[1][o] = (v1 < 0.f) ? 0.2f*v1 : v1;
            float v2 = as[s*Hq+2] + ad2; ev[2][o] = (v2 < 0.f) ? 0.2f*v2 : v2;
            float v3 = as[s*Hq+3] + ad3; ev[3][o] = (v3 < 0.f) ? 0.2f*v3 : v3;
            o++;
        }
    }
    __syncthreads();

    // pass 3: stable softmax per head (serial over <= ~40 neighbors)
    if (t < Hq) {
        float m = -3.402823466e38f;
        for (int i = 0; i < total; ++i) m = fmaxf(m, ev[t][i]);
        float den = 0.f;
        for (int i = 0; i < total; ++i) { float ex = expf(ev[t][i] - m); ev[t][i] = ex; den += ex; }
        float r = 1.f / den;
        for (int i = 0; i < total; ++i) ev[t][i] *= r;
    }
    __syncthreads();

    // pass 4: out[dst, dim] = GELU( sum_i att[h(dim)][i] * h[src_i, dim] + b[dim] )
    for (int dim = t; dim < Dq; dim += 256) {
        const int h = dim / DHq;
        float acc = 0.f;
        for (int i = 0; i < total; ++i)
            acc = fmaf(ev[h][i], hmat[(size_t)srcs[i] * Dq + dim], acc);
        xout[(size_t)dst * Dq + dim] = gelu_exact(acc + bias[dim]);
    }
}

// ---------------- 5) fused embedding sum + CLS scatter + LayerNorm ----------
// grid = B*S blocks, 192 threads, float4 per thread
__global__ void embed_ln_kernel(const int* __restrict__ src,
                                const int* __restrict__ seg,
                                const int* __restrict__ type_ids,
                                const int* __restrict__ pos,
                                const float* __restrict__ word_table,
                                const float* __restrict__ pos_table,
                                const float* __restrict__ seg_table,
                                const float* __restrict__ type_table,
                                const float* __restrict__ gamma,
                                const float* __restrict__ beta,
                                const float* __restrict__ xbuf,
                                float* __restrict__ out) {
    __shared__ float red[12];
    const int r = blockIdx.x;
    const int b = r >> 9;            // r / S
    const int s = r & (Sq - 1);      // r % S
    const int t = threadIdx.x;
    const int lane = t & 31, wid = t >> 5;

    const int tok = src[r];
    const float4* base4 = (s == 0)
        ? (const float4*)(xbuf + (size_t)(b * NPERq) * Dq)                // GNN text node
        : (const float4*)(word_table + (size_t)tok * Dq);
    const float4* p4  = (const float4*)(pos_table  + (size_t)pos[r]      * Dq);
    const float4* sg4 = (const float4*)(seg_table  + (size_t)seg[r]      * Dq);
    const float4* ty4 = (const float4*)(type_table + (size_t)type_ids[r] * Dq);

    float4 a = base4[t], pp = p4[t], ss = sg4[t], tt = ty4[t];
    float vx = a.x + pp.x + ss.x + tt.x;
    float vy = a.y + pp.y + ss.y + tt.y;
    float vz = a.z + pp.z + ss.z + tt.z;
    float vw = a.w + pp.w + ss.w + tt.w;

    float sum = vx + vy + vz + vw;
    float sq  = vx*vx + vy*vy + vz*vz + vw*vw;
    #pragma unroll
    for (int off = 16; off > 0; off >>= 1) {
        sum += __shfl_xor(sum, off, 32);
        sq  += __shfl_xor(sq,  off, 32);
    }
    if (lane == 0) { red[wid] = sum; red[6 + wid] = sq; }
    __syncthreads();
    float tot = 0.f, totq = 0.f;
    #pragma unroll
    for (int i = 0; i < 6; ++i) { tot += red[i]; totq += red[6 + i]; }

    const float mu  = tot * (1.f / Dq);
    const float var = totq * (1.f / Dq) - mu * mu;
    const float inv = rsqrtf(var + 1e-6f);

    const float4 g  = ((const float4*)gamma)[t];
    const float4 be = ((const float4*)beta)[t];
    float4 o;
    o.x = g.x * (vx - mu) * inv + be.x;
    o.y = g.y * (vy - mu) * inv + be.y;
    o.z = g.z * (vz - mu) * inv + be.z;
    o.w = g.w * (vw - mu) * inv + be.w;
    ((float4*)out)[(size_t)r * (Dq/4) + t] = o;
}

// ---------------------------- host launcher ----------------------------
extern "C" void kernel_launch(void* const* d_in, const int* in_sizes, int n_in,
                              void* d_out, int out_size, void* d_ws, size_t ws_size,
                              hipStream_t stream) {
    const int*   src        = (const int*)  d_in[0];
    const int*   seg        = (const int*)  d_in[1];
    const int*   type_ids   = (const int*)  d_in[2];
    const int*   pos        = (const int*)  d_in[3];
    const int*   span_start = (const int*)  d_in[4];
    const int*   span_end   = (const int*)  d_in[5];
    const int*   edge_idx   = (const int*)  d_in[6];
    const float* word_table = (const float*)d_in[7];
    const float* pos_table  = (const float*)d_in[8];
    const float* seg_table  = (const float*)d_in[9];
    const float* type_table = (const float*)d_in[10];
    const float* gamma      = (const float*)d_in[11];
    const float* beta       = (const float*)d_in[12];
    const float* W_gat      = (const float*)d_in[13];   // [5, D, D]
    const float* a_src      = (const float*)d_in[14];   // [5, H, DH]
    const float* a_dst      = (const float*)d_in[15];   // [5, H, DH]
    const float* b_gat      = (const float*)d_in[16];   // [5, D]
    float* out = (float*)d_out;

    // workspace layout (all f32): x[N*D], h[N*D], as[N*H], ad[N*H]  (~3.6 MB)
    float* xbuf = (float*)d_ws;
    float* hbuf = xbuf + (size_t)Nq * Dq;
    float* as   = hbuf + (size_t)Nq * Dq;
    float* ad   = as   + (size_t)Nq * Hq;

    assemble_nodes_kernel<<<Nq, 192, 0, stream>>>(src, span_start, span_end,
                                                  word_table, xbuf);

    for (int l = 0; l < 5; ++l) {
        const float* Wl  = W_gat + (size_t)l * Dq * Dq;
        const float* asl = a_src + (size_t)l * Hq * DHq;
        const float* adl = a_dst + (size_t)l * Hq * DHq;
        const float* bl  = b_gat + (size_t)l * Dq;

        gemm_wmma_kernel<<<dim3(Nq / 16, Dq / 128), 256, 0, stream>>>(xbuf, Wl, hbuf);
        gat_alpha_kernel<<<Nq, 128, 0, stream>>>(hbuf, asl, adl, as, ad);
        gat_aggregate_kernel<<<Nq, 256, 0, stream>>>(hbuf, as, ad, edge_idx, bl, xbuf);
    }

    embed_ln_kernel<<<Bq * Sq, 192, 0, stream>>>(src, seg, type_ids, pos,
                                                 word_table, pos_table, seg_table,
                                                 type_table, gamma, beta, xbuf, out);
}
